// BasicTransformerBlock_67370857005265
// MI455X (gfx1250) — compile-verified, matrix operand order checked by reading, not run on hardware
//
#include <hip/hip_runtime.h>
#include <hip/hip_bf16.h>

typedef __bf16 bf16_t;
typedef __attribute__((ext_vector_type(16))) __bf16 v16bf;
typedef __attribute__((ext_vector_type(8)))  float  v8f;
typedef __attribute__((ext_vector_type(4)))  unsigned u32x4;
typedef __attribute__((ext_vector_type(8)))  int      i32x8;
typedef __attribute__((ext_vector_type(4)))  int      i32x4;

#define WMMA_BF16(a, b, c) \
  __builtin_amdgcn_wmma_f32_16x16x32_bf16(false, (a), false, (b), (short)0, (c), false, false)

#if defined(__has_builtin)
#if __has_builtin(__builtin_amdgcn_tensor_load_to_lds) && \
    __has_builtin(__builtin_amdgcn_s_wait_tensorcnt)
#define USE_TDM 1
#else
#define USE_TDM 0
#endif
#else
#define USE_TDM 0
#endif

#define NTOK    4096
#define DMODEL  1152
#define NHEAD   16
#define DH      72
#define DHP     96      // dh padded to 3 * 32 for bf16 WMMA k-steps
#define KRETAIN 1638    // int(4096 * 0.4)
#define NPRUNE  (NTOK - KRETAIN)

static __device__ __forceinline__ bf16_t f2bf(float f) {
  unsigned u = __builtin_bit_cast(unsigned, f);
  unsigned r = u + 0x7FFFu + ((u >> 16) & 1u);   // round-to-nearest-even
  unsigned short hs = (unsigned short)(r >> 16);
  return __builtin_bit_cast(bf16_t, hs);
}

static __device__ __forceinline__ float xmax16(float v) {
  v = fmaxf(v, __shfl_xor(v, 1, 32));
  v = fmaxf(v, __shfl_xor(v, 2, 32));
  v = fmaxf(v, __shfl_xor(v, 4, 32));
  v = fmaxf(v, __shfl_xor(v, 8, 32));
  return v;
}
static __device__ __forceinline__ float xsum16(float v) {
  v += __shfl_xor(v, 1, 32);
  v += __shfl_xor(v, 2, 32);
  v += __shfl_xor(v, 4, 32);
  v += __shfl_xor(v, 8, 32);
  return v;
}

// ---------------------------------------------------------------- converts
__global__ void cvt_f32_bf16(const float* __restrict__ in, bf16_t* __restrict__ out, int n) {
  int i = blockIdx.x * blockDim.x + threadIdx.x;
  if (i < n) out[i] = f2bf(in[i]);
}

// transpose-convert: in R x C (f32, row-major) -> out C x R (bf16)
__global__ void cvt_transpose(const float* __restrict__ in, bf16_t* __restrict__ out,
                              int R, int C) {
  int c = blockIdx.x * 32 + (threadIdx.x & 31);
  int r = blockIdx.y * 8 + (threadIdx.x >> 5);
  if (c < C && r < R) out[(size_t)c * R + r] = f2bf(in[(size_t)r * C + c]);
}

// ------------------------------------------------- WMMA GEMM: C = A(MxK) @ B
// Bt is B TRANSPOSED (N x K, row-major) so every operand is one contiguous 32B load.
// mode 0: write f32 C (row-major, ld=N) with optional bias
// mode 1: write bf16 into per-head padded layout [head][row][DHP]
// mode 2: write bf16 into per-head TRANSPOSED layout [head][DHP][NTOK] (for V)
__global__ void gemm_bf16t(const bf16_t* __restrict__ A, const bf16_t* __restrict__ Bt,
                           int M, int N, int K,
                           float* __restrict__ Cf, const float* __restrict__ bias,
                           bf16_t* __restrict__ Ch, int mode) {
  const int tid = threadIdx.x, wave = tid >> 5, lane = tid & 31;
  const int mrow = lane & 15, khalf = (lane >> 4) * 16;
  const int row0 = blockIdx.y * 16;
  const int col0 = blockIdx.x * 128 + wave * 16;

  const bf16_t* ap0 = A + (size_t)(row0 + mrow) * K + khalf;
  const bf16_t* bp0 = Bt + (size_t)(col0 + mrow) * K + khalf;

  v8f acc;
  for (int r = 0; r < 8; ++r) acc[r] = 0.0f;

  for (int k0 = 0; k0 < K; k0 += 32) {
    v16bf a = *(const v16bf*)(ap0 + k0);
    v16bf b = *(const v16bf*)(bp0 + k0);
    __builtin_prefetch(ap0 + k0 + 64, 0, 1);   // global_prefetch_b8
    __builtin_prefetch(bp0 + k0 + 64, 0, 1);
    acc = WMMA_BF16(a, b, acc);
  }

  for (int r = 0; r < 8; ++r) {
    int m = r + 8 * (lane >> 4);
    int c = col0 + mrow;
    float v = acc[r];
    if (mode == 0) {
      if (bias) v += bias[c];
      Cf[(size_t)(row0 + m) * N + c] = v;
    } else if (mode == 1) {
      int hh = c / DH, wi = c % DH;
      Ch[((size_t)hh * NTOK + row0 + m) * DHP + wi] = f2bf(v);
    } else {
      int hh = c / DH, wi = c % DH;
      Ch[((size_t)hh * DHP + wi) * NTOK + (row0 + m)] = f2bf(v);
    }
  }
}

// --------------------------------------------- fused attention + head-mean P
// grid.x = NTOK/16 row blocks, 256 threads (8 waves), dynamic LDS = 16*4096*4 bytes
__global__ void attention_kernel(const bf16_t* __restrict__ Qh, const bf16_t* __restrict__ Kh,
                                 const bf16_t* __restrict__ Vt,
                                 float* __restrict__ attn, float* __restrict__ O) {
  extern __shared__ float accP[];                      // [16][4096] mean-P accumulator (256 KB)
  __shared__ float stats[8][16][2];                    // per-wave (max, sumexp) per row
  __shared__ float gmaxs[16], ginvs[16];
  __shared__ float opart[16][DHP];                     // cross-wave O reduction
  __shared__ __align__(32) bf16_t pstage[8][16][32];   // per-wave P tile (bf16 A operand)
  __shared__ __align__(32) bf16_t qstage[16][DHP];     // TDM-landed Q tile

  const int row0 = blockIdx.x * 16;
  const int tid = threadIdx.x, wave = tid >> 5, lane = tid & 31;
  const int mrow = lane & 15, khalf = (lane >> 4) * 16;
  const float rs = 0.117851130198f;                    // 72^-0.5

  for (int i = tid; i < 16 * NTOK; i += 256) accP[i] = 0.0f;
  __syncthreads();

  for (int h = 0; h < NHEAD; ++h) {
    const bf16_t* Q = Qh + ((size_t)h * NTOK + row0) * DHP;
    const bf16_t* K = Kh + (size_t)h * NTOK * DHP;
    const bf16_t* V = Vt + (size_t)h * DHP * NTOK;

    // ---- stage the 16x96 Q tile into LDS via the Tensor Data Mover
    v16bf aq[3];
#if USE_TDM
    if (wave == 0) {
      unsigned long long ga = (unsigned long long)(const void*)Q;
      unsigned la = (unsigned)(size_t)(&qstage[0][0]);
      u32x4 g0;
      g0[0] = 1u;                                         // count=1 (valid), user mode
      g0[1] = la;                                         // lds_addr
      g0[2] = (unsigned)(ga & 0xFFFFFFFFu);               // global_addr[31:0]
      g0[3] = (unsigned)((ga >> 32) & 0x01FFFFFFu)        // global_addr[56:32]
              | (2u << 30);                               // type = 2 ("image")
      i32x8 g1;
      g1[0] = (int)(1u << 16);              // data_size = 1 -> 2 bytes/elem
      g1[1] = (int)((unsigned)DHP << 16);   // tensor_dim0 = 96 (bits 79:48, low part)
      g1[2] = (int)(16u << 16);             // tensor_dim1 = 16 (bits 111:80)
      g1[3] = (int)((unsigned)DHP << 16);   // tile_dim0 = 96 (bits 127:112)
      g1[4] = 16;                           // tile_dim1 = 16 (bits 143:128)
      g1[5] = DHP;                          // tensor_dim0_stride = 96 (bits 207:160 low)
      g1[6] = 0;
      g1[7] = 0;
      i32x4 z4;
      z4[0] = 0; z4[1] = 0; z4[2] = 0; z4[3] = 0;
      i32x8 z8;
      for (int zz = 0; zz < 8; ++zz) z8[zz] = 0;
      // 6-arg form (clang-23 / therock-10.0 lane): groups 0..3 + extra group + cpol
      __builtin_amdgcn_tensor_load_to_lds(g0, g1, z4, z4, z8, 0);
    }
    __builtin_amdgcn_s_wait_tensorcnt(0);
    __syncthreads();
    for (int kk = 0; kk < 3; ++kk)
      aq[kk] = *(const v16bf*)(&qstage[mrow][kk * 32 + khalf]);
#else
    for (int kk = 0; kk < 3; ++kk)
      aq[kk] = *(const v16bf*)(Q + (size_t)mrow * DHP + kk * 32 + khalf);
    (void)qstage;
#endif

    // -------- pass 1: per-wave online row max / sumexp over its column tiles
    float rm[8], rsm[8];
    for (int r = 0; r < 8; ++r) { rm[r] = -1e30f; rsm[r] = 0.0f; }
    for (int jt = wave; jt < NTOK / 16; jt += 8) {
      int col0 = jt * 16;
      v8f s;
      for (int r = 0; r < 8; ++r) s[r] = 0.0f;
      for (int kk = 0; kk < 3; ++kk) {
        v16bf b = *(const v16bf*)(K + (size_t)(col0 + mrow) * DHP + kk * 32 + khalf);
        s = WMMA_BF16(aq[kk], b, s);
      }
      __builtin_prefetch(K + (size_t)(col0 + 128 + mrow) * DHP + khalf, 0, 1);
      for (int r = 0; r < 8; ++r) {
        float v = s[r] * rs;
        float tm = xmax16(v);
        float nm = fmaxf(rm[r], tm);
        float ts = xsum16(__expf(v - nm));
        rsm[r] = rsm[r] * __expf(rm[r] - nm) + ts;
        rm[r] = nm;
      }
    }
    if ((lane & 15) == 0) {
      for (int r = 0; r < 8; ++r) {
        int m = r + 8 * (lane >> 4);
        stats[wave][m][0] = rm[r];
        stats[wave][m][1] = rsm[r];
      }
    }
    __syncthreads();
    if (tid < 16) {
      float gm = -1e30f;
      for (int w = 0; w < 8; ++w) gm = fmaxf(gm, stats[w][tid][0]);
      float gs = 0.0f;
      for (int w = 0; w < 8; ++w) gs += stats[w][tid][1] * __expf(stats[w][tid][0] - gm);
      gmaxs[tid] = gm;
      ginvs[tid] = 1.0f / gs;
    }
    for (int i = tid; i < 16 * DHP; i += 256) (&opart[0][0])[i] = 0.0f;
    __syncthreads();

    // -------- pass 2: recompute S, emit mean-P to LDS, P@V via WMMA
    float gmv[8], giv[8];
    for (int r = 0; r < 8; ++r) {
      int m = r + 8 * (lane >> 4);
      gmv[r] = gmaxs[m];
      giv[r] = ginvs[m];
    }
    v8f oacc[6];
    for (int t = 0; t < 6; ++t)
      for (int r = 0; r < 8; ++r) oacc[t][r] = 0.0f;

    for (int jc = wave; jc < NTOK / 32; jc += 8) {
      int colbase = jc * 32;
      for (int half = 0; half < 2; ++half) {
        int col0 = colbase + half * 16;
        v8f s;
        for (int r = 0; r < 8; ++r) s[r] = 0.0f;
        for (int kk = 0; kk < 3; ++kk) {
          v16bf b = *(const v16bf*)(K + (size_t)(col0 + mrow) * DHP + kk * 32 + khalf);
          s = WMMA_BF16(aq[kk], b, s);
        }
        for (int r = 0; r < 8; ++r) {
          int m = r + 8 * (lane >> 4);
          float p = __expf(s[r] * rs - gmv[r]) * giv[r];
          accP[m * NTOK + col0 + mrow] += p * 0.0625f;   // mean over 16 heads
          pstage[wave][m][half * 16 + mrow] = f2bf(p);
        }
      }
      // A operand = P tile (16x32) from wave-private LDS stage (contiguous b128)
      v16bf ap = *(const v16bf*)&pstage[wave][mrow][khalf];
      // B operand = V^T rows: one contiguous 32B load per lane
      for (int t = 0; t < 6; ++t) {
        v16bf bv = *(const v16bf*)(V + ((size_t)(t * 16 + mrow)) * NTOK + colbase + khalf);
        oacc[t] = WMMA_BF16(ap, bv, oacc[t]);
      }
    }
    // cross-wave O reduction in LDS (ds_add_f32)
    for (int t = 0; t < 6; ++t)
      for (int r = 0; r < 8; ++r) {
        int m = r + 8 * (lane >> 4);
        atomicAdd(&opart[m][t * 16 + mrow], oacc[t][r]);
      }
    __syncthreads();
    for (int i = tid; i < 16 * DH; i += 256) {
      int m = i / DH, c = i % DH;
      O[(size_t)(row0 + m) * DMODEL + h * DH + c] = opart[m][c];
    }
    __syncthreads();
  }

  for (int i = tid; i < 16 * NTOK; i += 256) {
    int m = i >> 12, c = i & (NTOK - 1);
    attn[(size_t)(row0 + m) * NTOK + c] = accP[i];
  }
}

// ----------------------------------------------- PageRank power iteration
__global__ void init_dist(float* __restrict__ d) {
  int i = blockIdx.x * blockDim.x + threadIdx.x;
  if (i < NTOK) d[i] = 1.0f / (float)NTOK;
}

__global__ void spmv_kernel(const float* __restrict__ dist, const float* __restrict__ attn,
                            float* __restrict__ nxt) {
  int j = blockIdx.x * 256 + threadIdx.x;
  int i0 = blockIdx.y * 256;
  float s = 0.0f;
  for (int i = 0; i < 256; ++i)
    s += dist[i0 + i] * attn[(size_t)(i0 + i) * NTOK + j];
  atomicAdd(&nxt[j], s);
}

// ----------------------------------------------- top-k bookkeeping
__global__ void rank_kernel(const float* __restrict__ imp, int* __restrict__ retained) {
  int i = blockIdx.x * blockDim.x + threadIdx.x;
  if (i >= NTOK) return;
  float vi = imp[i];
  int c = 0;
  for (int j = 0; j < NTOK; ++j) {
    float vj = imp[j];
    c += (vj > vi) || (vj == vi && j < i);
  }
  retained[i] = (c < KRETAIN) ? 1 : 0;
}

__global__ void select_kernel(const int* __restrict__ retained,
                              int* __restrict__ impIdx, int* __restrict__ pruneIdx,
                              float* __restrict__ outImp, float* __restrict__ outPrune) {
  int i = blockIdx.x * blockDim.x + threadIdx.x;
  if (i >= NTOK) return;
  int cr = 0;
  for (int j = 0; j < i; ++j) cr += retained[j];
  if (retained[i]) {
    impIdx[cr] = i;
    outImp[cr] = (float)i;
  } else {
    int p = i - cr;
    pruneIdx[p] = i;
    outPrune[p] = (float)i;
  }
}

__global__ void argmax_kernel(const float* __restrict__ attn, const int* __restrict__ impIdx,
                              int* __restrict__ maxind) {
  int j = blockIdx.x * blockDim.x + threadIdx.x;
  if (j >= NTOK) return;
  float best = -1e30f;
  int bi = 0;
  for (int r = 0; r < KRETAIN; ++r) {
    float v = attn[(size_t)impIdx[r] * NTOK + j];
    if (v > best) { best = v; bi = r; }
  }
  maxind[j] = bi;
}

__global__ void finish_kernel(const float* __restrict__ dist, const int* __restrict__ maxind,
                              const int* __restrict__ pruneIdx,
                              float* __restrict__ outImportance, float* __restrict__ outSim) {
  int t = blockIdx.x * blockDim.x + threadIdx.x;
  if (t < NTOK) outImportance[t] = dist[t];
  if (t < NPRUNE) outSim[t] = (float)maxind[pruneIdx[t]];
}

// ----------------------------------------------------------------- launcher
extern "C" void kernel_launch(void* const* d_in, const int* in_sizes, int n_in,
                              void* d_out, int out_size, void* d_ws, size_t ws_size,
                              hipStream_t stream) {
  (void)in_sizes; (void)n_in; (void)out_size; (void)ws_size;
  const float* x  = (const float*)d_in[0];
  const float* Wq = (const float*)d_in[1];
  const float* Wk = (const float*)d_in[2];
  const float* Wv = (const float*)d_in[3];
  const float* Wo = (const float*)d_in[4];
  const float* bo = (const float*)d_in[5];
  float* out = (float*)d_out;

  char* ws = (char*)d_ws;
  size_t off = 0;
  auto take = [&](size_t bytes) -> char* {
    char* p = ws + off;
    off = (off + bytes + 255) & ~(size_t)255;
    return p;
  };

  bf16_t* xb   = (bf16_t*)take((size_t)NTOK * DMODEL * 2);
  bf16_t* wqbt = (bf16_t*)take((size_t)DMODEL * DMODEL * 2);  // transposed bf16 weights
  bf16_t* wkbt = (bf16_t*)take((size_t)DMODEL * DMODEL * 2);
  bf16_t* wvbt = (bf16_t*)take((size_t)DMODEL * DMODEL * 2);
  bf16_t* wobt = (bf16_t*)take((size_t)DMODEL * DMODEL * 2);
  bf16_t* Qh   = (bf16_t*)take((size_t)NHEAD * NTOK * DHP * 2);
  bf16_t* Kh   = (bf16_t*)take((size_t)NHEAD * NTOK * DHP * 2);
  bf16_t* Vt   = (bf16_t*)take((size_t)NHEAD * DHP * NTOK * 2);
  float*  attn = (float*)take((size_t)NTOK * NTOK * 4);
  float*  Obuf = (float*)take((size_t)NTOK * DMODEL * 4);
  bf16_t* Obf  = (bf16_t*)take((size_t)NTOK * DMODEL * 2);
  float*  dist = (float*)take((size_t)6 * NTOK * 4);
  int* retained = (int*)take((size_t)NTOK * 4);
  int* impIdx   = (int*)take((size_t)KRETAIN * 4);
  int* pruneIdx = (int*)take((size_t)NPRUNE * 4);
  int* maxind   = (int*)take((size_t)NTOK * 4);

  // output layout (floats): out | importance | imp_sorted | prune_sorted | sim_inds
  float* outImportance = out + (size_t)NTOK * DMODEL;
  float* outImpSorted  = outImportance + NTOK;
  float* outPrune      = outImpSorted + KRETAIN;
  float* outSim        = outPrune + NPRUNE;

  const int nx = NTOK * DMODEL;
  cvt_f32_bf16<<<(nx + 255) / 256, 256, 0, stream>>>(x, xb, nx);
  dim3 tg(DMODEL / 32, DMODEL / 8);
  cvt_transpose<<<tg, 256, 0, stream>>>(Wq, wqbt, DMODEL, DMODEL);
  cvt_transpose<<<tg, 256, 0, stream>>>(Wk, wkbt, DMODEL, DMODEL);
  cvt_transpose<<<tg, 256, 0, stream>>>(Wv, wvbt, DMODEL, DMODEL);
  cvt_transpose<<<tg, 256, 0, stream>>>(Wo, wobt, DMODEL, DMODEL);

  // zero padded head buffers so cols [72,96) stay zero
  (void)hipMemsetAsync(Qh, 0, (size_t)NHEAD * NTOK * DHP * 2, stream);
  (void)hipMemsetAsync(Kh, 0, (size_t)NHEAD * NTOK * DHP * 2, stream);
  (void)hipMemsetAsync(Vt, 0, (size_t)NHEAD * DHP * NTOK * 2, stream);

  dim3 gg(DMODEL / 128, NTOK / 16);
  gemm_bf16t<<<gg, 256, 0, stream>>>(xb, wqbt, NTOK, DMODEL, DMODEL, nullptr, nullptr, Qh, 1);
  gemm_bf16t<<<gg, 256, 0, stream>>>(xb, wkbt, NTOK, DMODEL, DMODEL, nullptr, nullptr, Kh, 1);
  gemm_bf16t<<<gg, 256, 0, stream>>>(xb, wvbt, NTOK, DMODEL, DMODEL, nullptr, nullptr, Vt, 2);

  attention_kernel<<<NTOK / 16, 256, 16 * NTOK * sizeof(float), stream>>>(Qh, Kh, Vt, attn, Obuf);

  (void)hipMemsetAsync(dist, 0, (size_t)6 * NTOK * 4, stream);
  init_dist<<<NTOK / 256, 256, 0, stream>>>(dist);
  for (int it = 0; it < 5; ++it)
    spmv_kernel<<<dim3(NTOK / 256, NTOK / 256), 256, 0, stream>>>(dist + (size_t)it * NTOK, attn,
                                                                  dist + (size_t)(it + 1) * NTOK);
  const float* imp = dist + (size_t)5 * NTOK;
  rank_kernel<<<NTOK / 256, 256, 0, stream>>>(imp, retained);
  select_kernel<<<NTOK / 256, 256, 0, stream>>>(retained, impIdx, pruneIdx, outImpSorted, outPrune);
  argmax_kernel<<<NTOK / 256, 256, 0, stream>>>(attn, impIdx, maxind);
  finish_kernel<<<NTOK / 256, 256, 0, stream>>>(imp, maxind, pruneIdx, outImportance, outSim);

  cvt_f32_bf16<<<(nx + 255) / 256, 256, 0, stream>>>(Obuf, Obf, nx);
  gemm_bf16t<<<gg, 256, 0, stream>>>(Obf, wobt, NTOK, DMODEL, DMODEL, out, bo, nullptr, 0);
}
